// FF_attention_module_17291538333717
// MI455X (gfx1250) — matrix operand order repaired and checked by reading, not compiled, not run on hardware
//
#include <hip/hip_runtime.h>
#include <hip/hip_bf16.h>
#include <cstdint>
#include <cstddef>

// ---------------------------------------------------------------------------
// Problem constants (from reference): B=1024, T=128, D=16, N=64, OUT=1
// ---------------------------------------------------------------------------
#define BB   1024
#define TT   128
#define DD   16
#define NN   64
#define DN   1024            // D*N
#define KRAW 1040            // D*(N+1) = gate GEMM K
#define KA   1056            // K padded to multiple of 32
#define KT   33              // KA/32 k-tiles
#define NTOT 3072            // 3 gates * DN
#define NT   192             // NTOT/16 n-tiles
#define MT   64              // B/16 m-tiles
#define GATE_BLOCKS 192      // (NT/8) * (MT/8) = 24 * 8
#define J_BLOCKS    64       // 4 m-chunks * 16 d

typedef __attribute__((ext_vector_type(16))) _Float16 v16h;
typedef __attribute__((ext_vector_type(8)))  float    v8f;

// ---------------------------------------------------------------------------
// Workspace layout (bytes). All sizes are multiples of 1024.
// ---------------------------------------------------------------------------
static constexpr size_t SZ_BPACK = (size_t)KT * NT * 32 * 16 * 2;      // gate weights, f16 frag
static constexpr size_t SZ_WJPK  = (size_t)DD * 2 * 4 * 32 * 16 * 2;   // W_j, f16 frag
static constexpr size_t SZ_X16   = (size_t)BB * TT * DD * 2;           // x in f16
static constexpr size_t SZ_GPRE  = (size_t)BB * NTOT * 4;              // gate preacts f32
static constexpr size_t SZ_JPRE  = (size_t)BB * DN * 4;                // j preacts f32
static constexpr size_t SZ_H     = (size_t)BB * DN * 4;                // h f32
static constexpr size_t SZ_APK   = (size_t)MT * KT * 32 * 16 * 2;      // A (inp) f16 frag
static constexpr size_t SZ_HJPK  = (size_t)DD * 2 * MT * 32 * 16 * 2;  // h f16 frag for j-GEMM
static constexpr size_t SZ_C     = (size_t)BB * DN * 4;
static constexpr size_t SZ_GACC  = (size_t)BB * DN * 4;
static constexpr size_t SZ_EACC  = (size_t)BB * DD * 4;

static constexpr size_t OFF_BPACK = 0;
static constexpr size_t OFF_WJPK  = OFF_BPACK + SZ_BPACK;
static constexpr size_t OFF_X16   = OFF_WJPK  + SZ_WJPK;
static constexpr size_t OFF_GPRE  = OFF_X16   + SZ_X16;
static constexpr size_t OFF_JPRE  = OFF_GPRE  + SZ_GPRE;
static constexpr size_t OFF_H     = OFF_JPRE  + SZ_JPRE;
static constexpr size_t OFF_APK   = OFF_H     + SZ_H;      // zero-span start
static constexpr size_t OFF_HJPK  = OFF_APK   + SZ_APK;
static constexpr size_t OFF_C     = OFF_HJPK  + SZ_HJPK;
static constexpr size_t OFF_GACC  = OFF_C     + SZ_C;
static constexpr size_t OFF_EACC  = OFF_GACC  + SZ_GACC;
static constexpr size_t ZERO_WORDS = (SZ_APK + SZ_HJPK + SZ_C + SZ_GACC + SZ_EACC) / 4;

__device__ __forceinline__ float sigmoidf_(float x) { return 1.0f / (1.0f + __expf(-x)); }

// A-fragment scatter position for element (row m, k-offset koff within a 32-wide
// k-tile), per ISA 7.12.2 "16-bit A-Matrix 16x32":
//   group = koff>>3 ; lane = (m&15) + 16*(group&1) ; half = (koff&7) + 8*(group>>1)
__device__ __forceinline__ void afrag_pos(int m, int koff, int& lane, int& half) {
    int grp = koff >> 3;
    lane = (m & 15) + 16 * (grp & 1);
    half = (koff & 7) + 8 * (grp >> 1);
}

// B-fragment (32xK x 16 cols): group = koff>>3 ; lane = (n&15) + 16*(group&1) ; half = (koff&7) + 8*(group>>1)
__device__ __forceinline__ void bfrag_pos(int n, int koff, int& lane, int& half) {
    int grp = koff >> 3;
    lane = (n & 15) + 16 * (grp & 1);
    half = (koff & 7) + 8 * (grp >> 1);
}

// ---------------------------------------------------------------------------
// Prep kernels
// ---------------------------------------------------------------------------
__global__ __launch_bounds__(256) void k_zero(uint32_t* p, size_t n) {
    size_t i = (size_t)blockIdx.x * blockDim.x + threadIdx.x;
    if (i < n) p[i] = 0u;
}

__global__ __launch_bounds__(256) void k_cvt_x(const float* __restrict__ x,
                                               _Float16* __restrict__ x16, int n) {
    int i = blockIdx.x * 256 + threadIdx.x;
    if (i < n) x16[i] = (_Float16)x[i];
}

// Pack Wi/Wf/Wo (each 1024x1040 f32, row-major) into one f16 B-fragment buffer
// covering N=3072, K=1056 (zero padded).  Layout: [kt][ntile][lane][16 halves].
__global__ __launch_bounds__(256) void k_pack_gates(const float* __restrict__ Wi,
                                                    const float* __restrict__ Wf,
                                                    const float* __restrict__ Wo,
                                                    _Float16* __restrict__ Bp) {
    int id = blockIdx.x * 256 + threadIdx.x;
    if (id >= NTOT * KA) return;
    int k  = id % KA;
    int nn = id / KA;
    int g   = nn >> 10;
    int row = nn & 1023;
    float v = 0.0f;
    if (k < KRAW) {
        const float* W = (g == 0) ? Wi : (g == 1) ? Wf : Wo;
        v = W[(size_t)row * KRAW + k];
    }
    int kt = k >> 5, koff = k & 31, lane, half;
    bfrag_pos(nn, koff, lane, half);
    Bp[(((size_t)kt * NT + (nn >> 4)) * 32 + lane) * 16 + half] = (_Float16)v;
}

// Pack W_j (16,64,64) into B-fragment layout: [d][kt(2)][ntile(4)][lane][16]
__global__ __launch_bounds__(256) void k_pack_wj(const float* __restrict__ Wj,
                                                 _Float16* __restrict__ Wp) {
    int id = blockIdx.x * 256 + threadIdx.x;
    if (id >= DD * NN * NN) return;
    int d = id >> 12, n = (id >> 6) & 63, m = id & 63;   // K index = n, N index = m
    int kt = n >> 5, koff = n & 31, lane, half;
    bfrag_pos(m, koff, lane, half);
    Wp[((((size_t)d * 2 + kt) * 4 + (m >> 4)) * 32 + lane) * 16 + half] = (_Float16)Wj[id];
}

// Write x_0 into A-pack columns 0..15 (h columns + pad already zeroed).
__global__ __launch_bounds__(256) void k_pack_x0(const float* __restrict__ x,
                                                 _Float16* __restrict__ Ap) {
    int id = blockIdx.x * 256 + threadIdx.x;
    if (id >= BB * DD) return;
    int b = id >> 4, d = id & 15;
    int lane, half;
    afrag_pos(b, d, lane, half);   // col k = d -> kt 0
    Ap[(((size_t)(b >> 4) * KT + 0) * 32 + lane) * 16 + half] =
        (_Float16)x[((size_t)b * TT + 0) * DD + d];
}

// ---------------------------------------------------------------------------
// Fused per-step GEMMs (one dispatch so the small j-GEMM overlaps the big
// gate GEMM instead of serializing behind it on the stream).
//
// Blocks [0, 192):   gate GEMM  C(1024x3072) = A(1024x1056) * B(1056x3072)
//                    block tile 128x128, 8 waves, wave tile 32x64.
// Blocks [192, 256): j GEMM     jpre[b, d*64+m] = sum_n h[b,d,n] * W_j[d][n][m]
//                    per d: M=1024, N=64, K=64; 8 waves, wave tile 32x64.
// ---------------------------------------------------------------------------
__global__ __launch_bounds__(256) void k_step_gemms(const v16h* __restrict__ Ap,
                                                    const v16h* __restrict__ Bp,
                                                    const v16h* __restrict__ Hp,
                                                    const v16h* __restrict__ Wp,
                                                    float* __restrict__ Gpre,
                                                    float* __restrict__ Jpre) {
    const int lane = threadIdx.x & 31;
    const int wid  = threadIdx.x >> 5;
    const int r0   = (lane >> 4) * 8;     // C frag: lanes 16-31 hold rows m0+8..15
    const int cn   = lane & 15;

    if (blockIdx.x < GATE_BLOCKS) {
        const int bn = blockIdx.x % (NT / 8);   // 0..23
        const int bm = blockIdx.x / (NT / 8);   // 0..7
        const int wm = wid & 3, wn = wid >> 2;
        const int mtb = bm * 8 + wm * 2;        // first of 2 m-tiles
        const int ntb = bn * 8 + wn * 4;        // first of 4 n-tiles

        v8f acc[2][4] = {};
        for (int kt = 0; kt < KT; ++kt) {
            v16h a0 = Ap[((size_t)(mtb + 0) * KT + kt) * 32 + lane];
            v16h a1 = Ap[((size_t)(mtb + 1) * KT + kt) * 32 + lane];
            v16h b0 = Bp[((size_t)kt * NT + ntb + 0) * 32 + lane];
            v16h b1 = Bp[((size_t)kt * NT + ntb + 1) * 32 + lane];
            v16h b2 = Bp[((size_t)kt * NT + ntb + 2) * 32 + lane];
            v16h b3 = Bp[((size_t)kt * NT + ntb + 3) * 32 + lane];
            if (kt + 1 < KT) {   // L2-resident, but hint the next tiles along
                __builtin_prefetch(&Ap[((size_t)(mtb + 0) * KT + kt + 1) * 32 + lane], 0, 1);
                __builtin_prefetch(&Bp[((size_t)(kt + 1) * NT + ntb) * 32 + lane], 0, 1);
            }
            acc[0][0] = __builtin_amdgcn_wmma_f32_16x16x32_f16(false, a0, false, b0, (short)0, acc[0][0], false, false);
            acc[0][1] = __builtin_amdgcn_wmma_f32_16x16x32_f16(false, a0, false, b1, (short)0, acc[0][1], false, false);
            acc[0][2] = __builtin_amdgcn_wmma_f32_16x16x32_f16(false, a0, false, b2, (short)0, acc[0][2], false, false);
            acc[0][3] = __builtin_amdgcn_wmma_f32_16x16x32_f16(false, a0, false, b3, (short)0, acc[0][3], false, false);
            acc[1][0] = __builtin_amdgcn_wmma_f32_16x16x32_f16(false, a1, false, b0, (short)0, acc[1][0], false, false);
            acc[1][1] = __builtin_amdgcn_wmma_f32_16x16x32_f16(false, a1, false, b1, (short)0, acc[1][1], false, false);
            acc[1][2] = __builtin_amdgcn_wmma_f32_16x16x32_f16(false, a1, false, b2, (short)0, acc[1][2], false, false);
            acc[1][3] = __builtin_amdgcn_wmma_f32_16x16x32_f16(false, a1, false, b3, (short)0, acc[1][3], false, false);
        }
        #pragma unroll
        for (int i = 0; i < 2; ++i) {
            int mbase = (mtb + i) * 16 + r0;
            #pragma unroll
            for (int j = 0; j < 4; ++j) {
                int col = (ntb + j) * 16 + cn;
                #pragma unroll
                for (int r = 0; r < 8; ++r)
                    Gpre[(size_t)(mbase + r) * NTOT + col] = acc[i][j][r];
            }
        }
    } else {
        const int id  = blockIdx.x - GATE_BLOCKS;  // 0..63
        const int d   = id >> 2;                   // 0..15
        const int mtb = (id & 3) * 16 + wid * 2;   // first of 2 m-tiles

        v8f acc[2][4] = {};
        #pragma unroll
        for (int kt = 0; kt < 2; ++kt) {
            v16h a0 = Hp[(((size_t)d * 2 + kt) * MT + mtb + 0) * 32 + lane];
            v16h a1 = Hp[(((size_t)d * 2 + kt) * MT + mtb + 1) * 32 + lane];
            #pragma unroll
            for (int j = 0; j < 4; ++j) {
                v16h bf = Wp[((((size_t)d * 2 + kt) * 4 + j) * 32 + lane)];
                acc[0][j] = __builtin_amdgcn_wmma_f32_16x16x32_f16(false, a0, false, bf, (short)0, acc[0][j], false, false);
                acc[1][j] = __builtin_amdgcn_wmma_f32_16x16x32_f16(false, a1, false, bf, (short)0, acc[1][j], false, false);
            }
        }
        #pragma unroll
        for (int i = 0; i < 2; ++i) {
            int mbase = (mtb + i) * 16 + r0;
            #pragma unroll
            for (int j = 0; j < 4; ++j) {
                int col = d * NN + j * 16 + cn;
                #pragma unroll
                for (int r = 0; r < 8; ++r)
                    Jpre[(size_t)(mbase + r) * DN + col] = acc[i][j][r];
            }
        }
    }
}

// ---------------------------------------------------------------------------
// Per-step elementwise update + streaming attention accumulation.
// One wave handles one (b,d): lanes cover n = {lane, lane+32}.
// g_n is accumulated as Gacc = sum_t e_t*h_t, Eacc = sum_t e_t  (softmax over t
// folded into a running ratio), so the (B,T,D,N) outputs tensor never exists.
// ---------------------------------------------------------------------------
__global__ __launch_bounds__(256) void k_update(int t,
        const float* __restrict__ x, const _Float16* __restrict__ x16,
        const float* __restrict__ Gpre, const float* __restrict__ Jpre,
        const float* __restrict__ Wi_b, const float* __restrict__ Wf_b,
        const float* __restrict__ Wo_b,
        const float* __restrict__ Uj,  const float* __restrict__ bj,
        const float* __restrict__ Fa,  const float* __restrict__ Fab,
        float* __restrict__ c, float* __restrict__ h,
        _Float16* __restrict__ Ap, _Float16* __restrict__ Hp,
        float* __restrict__ Gacc, float* __restrict__ Eacc) {
    const int lane = threadIdx.x & 31;
    const int gw   = blockIdx.x * 8 + (threadIdx.x >> 5);
    const int d = gw & 15, b = gw >> 4;
    const float xt = x[((size_t)b * TT + t) * DD + d];

    float hv[2];
    #pragma unroll
    for (int s = 0; s < 2; ++s) {
        int n  = lane + 32 * s;
        int dn = d * NN + n;
        size_t bi = (size_t)b * DN + dn;
        float gi = sigmoidf_(Gpre[(size_t)b * NTOT + dn]        + Wi_b[dn]);
        float gf = sigmoidf_(Gpre[(size_t)b * NTOT + 1024 + dn] + Wf_b[dn]);
        float go = sigmoidf_(Gpre[(size_t)b * NTOT + 2048 + dn] + Wo_b[dn]);
        float jv = tanhf(Jpre[bi] + xt * Uj[dn] + bj[dn]);
        float cv = c[bi] * gf + gi * jv;
        c[bi] = cv;
        float hh = go * tanhf(cv);
        h[bi] = hh;
        hv[s] = hh;
        _Float16 hf = (_Float16)hh;
        { // scatter into gate-GEMM A pack at column k = 16 + dn
            int k = 16 + dn, kt = k >> 5, koff = k & 31, ln, hi;
            afrag_pos(b, koff, ln, hi);
            Ap[(((size_t)(b >> 4) * KT + kt) * 32 + ln) * 16 + hi] = hf;
        }
        { // scatter into j-GEMM A pack (K index = n)
            int kt = n >> 5, koff = n & 31, ln, hi;
            afrag_pos(b, koff, ln, hi);
            Hp[((((size_t)d * 2 + kt) * MT + (b >> 4)) * 32 + ln) * 16 + hi] = hf;
        }
    }
    // streaming attention: e = exp(tanh(<h, Fa[d]> + Fab[d]))
    float part = hv[0] * Fa[d * NN + lane] + hv[1] * Fa[d * NN + lane + 32];
    #pragma unroll
    for (int off = 16; off >= 1; off >>= 1) part += __shfl_xor(part, off, 32);
    float e = __expf(tanhf(part + Fab[d]));
    #pragma unroll
    for (int s = 0; s < 2; ++s) {
        int dn = d * NN + lane + 32 * s;
        Gacc[(size_t)b * DN + dn] += e * hv[s];
    }
    if (lane == 0) {
        Eacc[b * DD + d] += e;
        if (t + 1 < TT) { // stage x_{t+1} into A pack cols 0..15
            int ln, hi;
            afrag_pos(b, d, ln, hi);
            Ap[(((size_t)(b >> 4) * KT + 0) * 32 + ln) * 16 + hi] =
                x16[((size_t)b * TT + (t + 1)) * DD + d];
        }
    }
}

// ---------------------------------------------------------------------------
// Final head: g = Gacc/Eacc ; hg = [g, h_T] ; mu = hg.Phi + b ;
// beta = softmax_d(exp(tanh(hg.Fb + b))) ; out[b] = sum_d beta*mu
// ---------------------------------------------------------------------------
__global__ __launch_bounds__(256) void k_final(const float* __restrict__ Gacc,
                                               const float* __restrict__ Eacc,
                                               const float* __restrict__ h,
                                               const float* __restrict__ Phiw,
                                               const float* __restrict__ Phib,
                                               const float* __restrict__ Fbw,
                                               const float* __restrict__ Fbb,
                                               float* __restrict__ out) {
    __shared__ float mu_s[DD], bp_s[DD];
    const int b = blockIdx.x;
    const int lane = threadIdx.x & 31;
    const int w = threadIdx.x >> 5;
    #pragma unroll
    for (int dd = 0; dd < 2; ++dd) {
        int d = w * 2 + dd;
        float einv = 1.0f / Eacc[b * DD + d];
        float mu = 0.0f, bp = 0.0f;
        #pragma unroll
        for (int q = 0; q < 4; ++q) {
            int k = lane + q * 32;                     // 0..127 over hg
            float hg = (k < NN) ? Gacc[(size_t)b * DN + d * NN + k] * einv
                                : h[(size_t)b * DN + d * NN + (k - NN)];
            mu += hg * Phiw[k];
            bp += hg * Fbw[k];
        }
        #pragma unroll
        for (int off = 16; off >= 1; off >>= 1) {
            mu += __shfl_xor(mu, off, 32);
            bp += __shfl_xor(bp, off, 32);
        }
        if (lane == 0) { mu_s[d] = mu + Phib[0]; bp_s[d] = bp + Fbb[0]; }
    }
    __syncthreads();
    if (threadIdx.x == 0) {
        float se = 0.0f, sm = 0.0f;
        #pragma unroll
        for (int d = 0; d < DD; ++d) {
            float be = __expf(tanhf(bp_s[d]));
            se += be; sm += be * mu_s[d];
        }
        out[b] = sm / se;
    }
}

// ---------------------------------------------------------------------------
extern "C" void kernel_launch(void* const* d_in, const int* in_sizes, int n_in,
                              void* d_out, int out_size, void* d_ws, size_t ws_size,
                              hipStream_t stream) {
    (void)in_sizes; (void)n_in; (void)out_size; (void)ws_size;
    const float* x    = (const float*)d_in[0];
    const float* U_j  = (const float*)d_in[1];
    const float* W_j  = (const float*)d_in[2];
    const float* b_j  = (const float*)d_in[3];
    const float* Wi_w = (const float*)d_in[4];
    const float* Wi_b = (const float*)d_in[5];
    const float* Wf_w = (const float*)d_in[6];
    const float* Wf_b = (const float*)d_in[7];
    const float* Wo_w = (const float*)d_in[8];
    const float* Wo_b = (const float*)d_in[9];
    const float* Fa   = (const float*)d_in[10];
    const float* Fab  = (const float*)d_in[11];
    const float* Fbw  = (const float*)d_in[12];
    const float* Fbb  = (const float*)d_in[13];
    const float* Phiw = (const float*)d_in[14];
    const float* Phib = (const float*)d_in[15];
    float* out = (float*)d_out;

    char* ws = (char*)d_ws;
    _Float16* Bpack = (_Float16*)(ws + OFF_BPACK);
    _Float16* Wjpk  = (_Float16*)(ws + OFF_WJPK);
    _Float16* x16   = (_Float16*)(ws + OFF_X16);
    float*    Gpre  = (float*)(ws + OFF_GPRE);
    float*    Jpre  = (float*)(ws + OFF_JPRE);
    float*    Hf32  = (float*)(ws + OFF_H);
    _Float16* Apack = (_Float16*)(ws + OFF_APK);
    _Float16* Hjpk  = (_Float16*)(ws + OFF_HJPK);
    float*    Cst   = (float*)(ws + OFF_C);
    float*    Gacc  = (float*)(ws + OFF_GACC);
    float*    Eacc  = (float*)(ws + OFF_EACC);

    // --- one-time prep (per launch; deterministic) ---
    k_pack_gates<<<(NTOT * KA + 255) / 256, 256, 0, stream>>>(Wi_w, Wf_w, Wo_w, Bpack);
    k_pack_wj<<<(DD * NN * NN + 255) / 256, 256, 0, stream>>>(W_j, Wjpk);
    k_cvt_x<<<(BB * TT * DD + 255) / 256, 256, 0, stream>>>(x, x16, BB * TT * DD);
    k_zero<<<(unsigned)((ZERO_WORDS + 255) / 256), 256, 0, stream>>>((uint32_t*)(ws + OFF_APK), ZERO_WORDS);
    k_pack_x0<<<(BB * DD + 255) / 256, 256, 0, stream>>>(x, Apack);

    // --- sequential scan over T: 2 dispatches per step (GEMMs fused) ---
    for (int t = 0; t < TT; ++t) {
        k_step_gemms<<<GATE_BLOCKS + J_BLOCKS, 256, 0, stream>>>(
            (const v16h*)Apack, (const v16h*)Bpack,
            (const v16h*)Hjpk,  (const v16h*)Wjpk, Gpre, Jpre);
        k_update<<<(BB * DD) / 8, 256, 0, stream>>>(t, x, x16, Gpre, Jpre,
                                                    Wi_b, Wf_b, Wo_b, U_j, b_j, Fa, Fab,
                                                    Cst, Hf32, Apack, Hjpk, Gacc, Eacc);
    }

    // --- head ---
    k_final<<<BB, 256, 0, stream>>>(Gacc, Eacc, Hf32, Phiw, Phib, Fbw, Fbb, out);
}
